// VAEModel_9612136808731
// MI455X (gfx1250) — compile-verified, matrix operand order checked by reading, not run on hardware
//
#include <hip/hip_runtime.h>

// ---------------------------------------------------------------------------
// CDNA5 (gfx1250) implementation. wave32.
//  - All GEMMs: v_wmma_f32_16x16x32_bf16, 128x64 block tile, 8 waves,
//    4 WMMAs/wave/K-step.
//  - A-tile staging for pure-fp32-activation GEMMs: Tensor Data Mover
//    (tensor_load_to_lds + s_wait_tensorcnt), fp32 image in LDS, bf16
//    conversion fused into the fragment load.
// ---------------------------------------------------------------------------

typedef __attribute__((ext_vector_type(16))) __bf16 v16bf;
typedef __attribute__((ext_vector_type(8)))  __bf16 v8bf;
typedef __attribute__((ext_vector_type(8)))  float  v8f;
typedef __attribute__((ext_vector_type(4)))  unsigned int u32x4;
typedef __attribute__((ext_vector_type(8)))  int i32x8;
typedef __attribute__((ext_vector_type(4)))  int i32x4;

#define E_DIM  1024
#define H_DIM  512
#define LD_DIM 512
#define B_DIM  2048
#define L_DIM  48

// ---------------------------------------------------------------------------
template <int ACT>
__device__ __forceinline__ float act_fn(float x) {
    if constexpr (ACT == 1) {          // leaky relu
        return x > 0.0f ? x : 0.2f * x;
    } else if constexpr (ACT == 2) {   // clip [-1, 1]
        return fminf(fmaxf(x, -1.0f), 1.0f);
    } else {
        return x;
    }
}

// 16x32 bf16 fragment row for this lane from bf16 LDS (row stride 48 bf16).
// Lane L: row = base + L%16; half hh = L/16.
//   elements 0..7  -> K = hh*8 + e
//   elements 8..15 -> K = 16 + hh*8 + (e-8)
__device__ __forceinline__ v16bf load_frag_bf(const __bf16* rowbase, int hh) {
    const v8bf lo = *(const v8bf*)(rowbase + hh * 8);
    const v8bf hi = *(const v8bf*)(rowbase + 16 + hh * 8);
    v16bf r;
#pragma unroll
    for (int i = 0; i < 8; ++i) { r[i] = lo[i]; r[i + 8] = hi[i]; }
    return r;
}

// Same fragment but from an fp32 LDS image (row stride 40 floats), converting
// to bf16 in-register (TDM staging path).
__device__ __forceinline__ v16bf load_frag_f32(const float* rowbase, int hh) {
    const float4 l0 = *(const float4*)(rowbase + hh * 8);
    const float4 l1 = *(const float4*)(rowbase + hh * 8 + 4);
    const float4 h0 = *(const float4*)(rowbase + 16 + hh * 8);
    const float4 h1 = *(const float4*)(rowbase + 16 + hh * 8 + 4);
    v16bf r;
    r[0]  = (__bf16)l0.x; r[1]  = (__bf16)l0.y; r[2]  = (__bf16)l0.z; r[3]  = (__bf16)l0.w;
    r[4]  = (__bf16)l1.x; r[5]  = (__bf16)l1.y; r[6]  = (__bf16)l1.z; r[7]  = (__bf16)l1.w;
    r[8]  = (__bf16)h0.x; r[9]  = (__bf16)h0.y; r[10] = (__bf16)h0.z; r[11] = (__bf16)h0.w;
    r[12] = (__bf16)h1.x; r[13] = (__bf16)h1.y; r[14] = (__bf16)h1.z; r[15] = (__bf16)h1.w;
    return r;
}

// ---------------------------------------------------------------------------
// Tiled WMMA GEMM:  C[M,N] = act( A @ W + bias )
//   AMODE 0: A = virtual concat of A0 (K0 cols) | A1 (K-K0 cols), fp32,
//            vector-staged to LDS as bf16.
//   AMODE 1: A = A0 only (lda = K0 = K), staged by the Tensor Data Mover as a
//            32x128 fp32 tile (row-padded 32->40 DWORDs), converted at
//            fragment load.
//   W: (K,N) row-major fp32, vector-staged transposed to Bs[n][k] bf16.
// Block tile 128x64, BK=32, 256 threads (8 waves). Wave (wm,wn2) owns rows
// [wm*32,+32) x cols [wn2*32,+32) -> 2x2 subtiles -> 4 WMMAs per K-step.
// ---------------------------------------------------------------------------
template <int ACT, int AMODE>
__global__ __launch_bounds__(256) void gemm_wmma(
    const float* __restrict__ A0, int K0,
    const float* __restrict__ A1, int K,
    const float* __restrict__ W,
    const float* __restrict__ bias,
    float* __restrict__ C, int M, int N) {

    constexpr int BM = 128, BN = 64, BK = 32;
    constexpr int STRB = 48;   // bf16 LDS row stride (96B, 16B aligned)
    constexpr int STRF = 40;   // fp32 LDS row stride (160B, 16B aligned)

    // A staging buffer: max(128*48*2, 128*40*4) = 20480 bytes
    __shared__ __align__(16) unsigned char AsRaw[BM * STRF * 4];
    __shared__ __bf16 Bs[BN * STRB];
    __bf16* Asb = (__bf16*)AsRaw;
    float*  Asf = (float*)AsRaw;

    const int tid  = threadIdx.x;
    const int wave = tid >> 5;
    const int lane = tid & 31;
    const int wm   = wave >> 1;   // 0..3 : 32-row stripe
    const int wn2  = wave & 1;    // 0..1 : 32-col stripe

    const int block_m = blockIdx.y * BM;
    const int block_n = blockIdx.x * BN;

    const int a_row = tid >> 2;          // 0..63 (two passes for 128 rows)
    const int a_k   = (tid & 3) << 3;    // 0,8,16,24
    const int b_n   = tid & 63;
    const int b_kb  = (tid >> 6) << 3;   // 0,8,16,24

    const int fr = lane & 15;
    const int hh = lane >> 4;

    v8f acc[2][2];
    acc[0][0] = {}; acc[0][1] = {}; acc[1][0] = {}; acc[1][1] = {};

    const unsigned lds_off = (unsigned)(size_t)AsRaw;

    for (int kk = 0; kk < K; kk += BK) {
        __syncthreads();

        if constexpr (AMODE == 1) {
            // ---- TDM: DMA a 32(K) x 128(rows) fp32 tile into LDS ----
            if (wave == 0) {
                const unsigned long long ga =
                    (unsigned long long)(size_t)(A0 + (size_t)block_m * (size_t)K0 + kk);
                u32x4 g0;
                g0[0] = 1u;                                  // count=1, user mode
                g0[1] = lds_off;                             // lds_addr
                g0[2] = (unsigned)(ga & 0xffffffffu);        // global_addr[31:0]
                g0[3] = (unsigned)((ga >> 32) & 0x1ffffffu)  // global_addr[56:32]
                        | (2u << 30);                        // type = image
                i32x8 g1;
                g1[0] = (int)((2u << 16)      // data_size = 4 bytes
                              | (1u << 20)    // pad_enable
                              | (4u << 22)    // pad_interval: 32 DWORDs
                              | (7u << 25));  // pad_amount:   8 DWORDs
                g1[1] = (int)(((unsigned)BK & 0xffffu) << 16);          // tensor_dim0 lo
                g1[2] = (int)(((unsigned)BK >> 16)
                              | (((unsigned)(M - block_m)) << 16));     // dim0 hi | dim1 lo
                g1[3] = (int)((((unsigned)(M - block_m)) >> 16)
                              | ((unsigned)BK << 16));                  // dim1 hi | tile_dim0
                g1[4] = (int)((unsigned)BM);                            // tile_dim1 | tile_dim2=0
                g1[5] = (int)((unsigned)K0);                            // tensor_dim0_stride lo32
                g1[6] = 0;
                g1[7] = 0;
                const i32x4 gz4 = {0, 0, 0, 0};
                const i32x8 gz8 = {0, 0, 0, 0, 0, 0, 0, 0};
                __builtin_amdgcn_tensor_load_to_lds(g0, g1, gz4, gz4, gz8, 0);
                __builtin_amdgcn_s_wait_tensorcnt(0);
            }
        } else {
            // ---- vector staging: fp32 -> bf16, 8 floats/thread, 2 passes ----
#pragma unroll
            for (int hpass = 0; hpass < 2; ++hpass) {
                const int row = a_row + hpass * 64;
                const float* src; int lda, col;
                const int gk = kk + a_k;
                if (gk < K0) { src = A0; lda = K0;     col = gk; }
                else         { src = A1; lda = K - K0; col = gk - K0; }
                const float* p = src + (size_t)(block_m + row) * (size_t)lda + col;
                const float4 f0 = *(const float4*)p;
                const float4 f1 = *(const float4*)(p + 4);
                v8bf h;
                h[0] = (__bf16)f0.x; h[1] = (__bf16)f0.y;
                h[2] = (__bf16)f0.z; h[3] = (__bf16)f0.w;
                h[4] = (__bf16)f1.x; h[5] = (__bf16)f1.y;
                h[6] = (__bf16)f1.z; h[7] = (__bf16)f1.w;
                *(v8bf*)(&Asb[row * STRB + a_k]) = h;
            }
        }

        // ---- stage B tile transposed: Bs[n][k] = W[kk+k][block_n+n] ----
        {
            const float* p = W + (size_t)(kk + b_kb) * (size_t)N + block_n + b_n;
            v8bf h;
#pragma unroll
            for (int j = 0; j < 8; ++j) h[j] = (__bf16)p[(size_t)j * (size_t)N];
            *(v8bf*)(&Bs[b_n * STRB + b_kb]) = h;
        }
        __syncthreads();

        // ---- fragments + 4 WMMAs ----
        v16bf a0, a1;
        if constexpr (AMODE == 1) {
            a0 = load_frag_f32(&Asf[(wm * 32 + fr) * STRF], hh);
            a1 = load_frag_f32(&Asf[(wm * 32 + 16 + fr) * STRF], hh);
        } else {
            a0 = load_frag_bf(&Asb[(wm * 32 + fr) * STRB], hh);
            a1 = load_frag_bf(&Asb[(wm * 32 + 16 + fr) * STRB], hh);
        }
        const v16bf b0 = load_frag_bf(&Bs[(wn2 * 32 + fr) * STRB], hh);
        const v16bf b1 = load_frag_bf(&Bs[(wn2 * 32 + 16 + fr) * STRB], hh);

        acc[0][0] = __builtin_amdgcn_wmma_f32_16x16x32_bf16(
            false, a0, false, b0, (short)0, acc[0][0], false, false);
        acc[0][1] = __builtin_amdgcn_wmma_f32_16x16x32_bf16(
            false, a0, false, b1, (short)0, acc[0][1], false, false);
        acc[1][0] = __builtin_amdgcn_wmma_f32_16x16x32_bf16(
            false, a1, false, b0, (short)0, acc[1][0], false, false);
        acc[1][1] = __builtin_amdgcn_wmma_f32_16x16x32_bf16(
            false, a1, false, b1, (short)0, acc[1][1], false, false);
    }

    // ---- epilogue: D (VGPR r, lane L) -> row = r + 8*(L/16), col = L%16 ----
#pragma unroll
    for (int sn = 0; sn < 2; ++sn) {
        const int col = block_n + wn2 * 32 + sn * 16 + fr;
        const float bv = bias[col];
#pragma unroll
        for (int sm = 0; sm < 2; ++sm) {
#pragma unroll
            for (int r = 0; r < 8; ++r) {
                const int row = block_m + wm * 32 + sm * 16 + r + hh * 8;
                C[(size_t)row * (size_t)N + col] = act_fn<ACT>(acc[sm][sn][r] + bv);
            }
        }
    }
}

// ---------------------------------------------------------------------------
// ctx[(b*48+l)*512 + d] = pair_embed[clip(start_b + l, 0, Nrows-1)*512 + d]
// ---------------------------------------------------------------------------
__global__ __launch_bounds__(256) void gather_ctx(
    const float* __restrict__ pe, const int* __restrict__ sse,
    float* __restrict__ ctx, int Nrows) {
    const size_t i = (size_t)blockIdx.x * 256 + threadIdx.x;
    const size_t total = (size_t)B_DIM * L_DIM * LD_DIM;
    if (i >= total) return;
    const int d  = (int)(i & (LD_DIM - 1));
    const size_t bl = i >> 9;
    const int l = (int)(bl % L_DIM);
    const int b = (int)(bl / L_DIM);
    long idx = (long)sse[2 * b] + l;
    if (idx < 0) idx = 0;
    if (idx > Nrows - 1) idx = Nrows - 1;
    ctx[i] = pe[(size_t)idx * LD_DIM + d];
}

// ---------------------------------------------------------------------------
// Per-batch attention -> seg_out. One 256-thread block per batch element.
// ---------------------------------------------------------------------------
__global__ __launch_bounds__(256) void attn_seg(
    const float* __restrict__ q, const float* __restrict__ k,
    const float* __restrict__ v, const int* __restrict__ sse,
    float* __restrict__ seg) {

    __shared__ float qs[L_DIM * 128];
    __shared__ float ks[L_DIM * 128];
    __shared__ float sc[L_DIM * L_DIM];
    __shared__ float cm[L_DIM];

    const int b   = blockIdx.x;
    const int tid = threadIdx.x;
    const int len = sse[2 * b + 1] - sse[2 * b];

    const float* qb = q + (size_t)b * L_DIM * LD_DIM;
    const float* kb = k + (size_t)b * L_DIM * LD_DIM;

    float accv[9];
#pragma unroll
    for (int e = 0; e < 9; ++e) accv[e] = 0.0f;

    for (int d0 = 0; d0 < LD_DIM; d0 += 128) {
        __syncthreads();
        for (int i = tid; i < L_DIM * 128; i += 256) {
            const int r = i >> 7, c = i & 127;
            qs[i] = qb[(size_t)r * LD_DIM + d0 + c];
            ks[i] = kb[(size_t)r * LD_DIM + d0 + c];
        }
        __syncthreads();
#pragma unroll
        for (int e = 0; e < 9; ++e) {
            const int idx = tid * 9 + e;               // 2304 = 256*9
            const int l = idx / L_DIM, m = idx % L_DIM;
            const float* ql = &qs[l * 128];
            const float* km = &ks[m * 128];
            float s = 0.0f;
            for (int c = 0; c < 128; ++c) s += ql[c] * km[c];
            accv[e] += s;
        }
    }
    __syncthreads();

    const float scale = 0.044194173824159216f;  // 1/sqrt(512)
#pragma unroll
    for (int e = 0; e < 9; ++e) {
        const int idx = tid * 9 + e;
        const int m = idx % L_DIM;
        float s = accv[e] * scale;
        if (m >= len) s = -1.0e9f;
        sc[idx] = s;
    }
    __syncthreads();

    if (tid < L_DIM) {  // softmax over row tid
        float mx = -1.0e30f;
        for (int m = 0; m < L_DIM; ++m) mx = fmaxf(mx, sc[tid * L_DIM + m]);
        float sum = 0.0f;
        for (int m = 0; m < L_DIM; ++m) {
            const float e = __expf(sc[tid * L_DIM + m] - mx);
            sc[tid * L_DIM + m] = e;
            sum += e;
        }
        const float inv = 1.0f / sum;
        for (int m = 0; m < L_DIM; ++m) sc[tid * L_DIM + m] *= inv;
    }
    __syncthreads();

    if (tid < L_DIM) {  // column sums over valid query rows
        float c = 0.0f;
        for (int l = 0; l < len; ++l) c += sc[l * L_DIM + tid];
        cm[tid] = c;
    }
    __syncthreads();

    const float* vb = v + (size_t)b * L_DIM * LD_DIM;
    const float invlen = 1.0f / (float)len;
    for (int d = tid; d < LD_DIM; d += 256) {
        float s = 0.0f;
        for (int m = 0; m < L_DIM; ++m) s += cm[m] * vb[(size_t)m * LD_DIM + d];
        seg[(size_t)b * LD_DIM + d] = s * invlen;
    }
}

// ---------------------------------------------------------------------------
// z = normalize(mean + exp(0.5*lv)*eps) * sqrt(LD). One block per row.
// ---------------------------------------------------------------------------
__global__ __launch_bounds__(256) void z_kernel(
    const float* __restrict__ mean, const float* __restrict__ lv,
    const float* __restrict__ eps, float* __restrict__ z) {
    __shared__ float red[256];
    const int b = blockIdx.x, tid = threadIdx.x;
    const size_t base = (size_t)b * LD_DIM;
    float tv[2];
    float local = 0.0f;
#pragma unroll
    for (int j = 0; j < 2; ++j) {
        const int d = tid + j * 256;
        const float t = mean[base + d] + __expf(0.5f * lv[base + d]) * eps[base + d];
        tv[j] = t;
        local += t * t;
    }
    red[tid] = local;
    __syncthreads();
    for (int s = 128; s > 0; s >>= 1) {
        if (tid < s) red[tid] += red[tid + s];
        __syncthreads();
    }
    const float nrm = fmaxf(sqrtf(red[0]), 1.0e-12f);
    const float scl = 22.627416997969522f / nrm;  // sqrt(512)
#pragma unroll
    for (int j = 0; j < 2; ++j) {
        const int d = tid + j * 256;
        z[base + d] = tv[j] * scl;
    }
}

// ---------------------------------------------------------------------------
// out[row] = dot(U[row,0:512], w) + bias[0]. One wave32 per row.
// ---------------------------------------------------------------------------
__global__ __launch_bounds__(256) void rowdot(
    const float* __restrict__ U, const float* __restrict__ w,
    const float* __restrict__ bias, float* __restrict__ out, int M) {
    const int row  = blockIdx.x * 8 + (threadIdx.x >> 5);
    const int lane = threadIdx.x & 31;
    if (row >= M) return;
    const float* u = U + (size_t)row * H_DIM;
    float s = 0.0f;
    for (int i = lane; i < H_DIM; i += 32) s += u[i] * w[i];
#pragma unroll
    for (int off = 16; off > 0; off >>= 1) s += __shfl_down(s, off, 32);
    if (lane == 0) out[row] = s + bias[0];
}

// ---------------------------------------------------------------------------
extern "C" void kernel_launch(void* const* d_in, const int* in_sizes, int n_in,
                              void* d_out, int out_size, void* d_ws, size_t ws_size,
                              hipStream_t stream) {
    const float* tc   = (const float*)d_in[0];
    const float* tr   = (const float*)d_in[1];
    const float* cc   = (const float*)d_in[2];
    const float* cr   = (const float*)d_in[3];
    const float* eps  = (const float*)d_in[4];
    const float* pw1  = (const float*)d_in[5];
    const float* pb1  = (const float*)d_in[6];
    const float* pw2  = (const float*)d_in[7];
    const float* pb2  = (const float*)d_in[8];
    const float* pw3  = (const float*)d_in[9];
    const float* pb3  = (const float*)d_in[10];
    const float* wq   = (const float*)d_in[11];
    const float* bq   = (const float*)d_in[12];
    const float* wk   = (const float*)d_in[13];
    const float* bk   = (const float*)d_in[14];
    const float* wv   = (const float*)d_in[15];
    const float* bv   = (const float*)d_in[16];
    const float* wm   = (const float*)d_in[17];
    const float* bm   = (const float*)d_in[18];
    const float* wlv  = (const float*)d_in[19];
    const float* blv  = (const float*)d_in[20];
    const float* dw1  = (const float*)d_in[21];
    const float* db1  = (const float*)d_in[22];
    const float* dw2  = (const float*)d_in[23];
    const float* db2  = (const float*)d_in[24];
    const float* dw3  = (const float*)d_in[25];
    const float* db3  = (const float*)d_in[26];
    const int*   sse  = (const int*)d_in[27];
    // d_in[28] = user_type (unused by the reference)

    const int Nctx = in_sizes[2] / E_DIM;   // 65536
    const int BL   = B_DIM * L_DIM;         // 98304

    // ---- workspace layout (floats) ----
    float* ws = (float*)d_ws;
    const size_t NS = (size_t)Nctx * H_DIM;
    const size_t CT = (size_t)BL * LD_DIM;
    float* h1   = ws;
    float* h2   = ws + NS;
    float* ctx  = ws + 2 * NS;
    float* qb   = ctx + CT;
    float* kb   = qb + CT;
    float* vb   = kb + CT;
    float* seg  = vb + CT;
    float* pe   = h1;   // pair_embed reuses h1
    float* u1   = qb;   // decoder hidden 1 reuses q
    float* u2   = kb;   // decoder hidden 2 reuses k

    // ---- output layout: rc, rr, mean, log_var, z ----
    float* out   = (float*)d_out;
    float* rc    = out;
    float* rr    = out + B_DIM;
    float* omean = out + 2 * B_DIM;
    float* olv   = omean + (size_t)B_DIM * LD_DIM;
    float* oz    = olv + (size_t)B_DIM * LD_DIM;

    const dim3 blk(256);

    // 1) pair-embed MLP (concat(cc,cr) -> 512 -> 512 -> 512)
    {
        const dim3 g(H_DIM / 64, Nctx / 128);
        gemm_wmma<1, 0><<<g, blk, 0, stream>>>(cc, E_DIM, cr, 2 * E_DIM, pw1, pb1, h1, Nctx, H_DIM);
        gemm_wmma<1, 1><<<g, blk, 0, stream>>>(h1, H_DIM, nullptr, H_DIM, pw2, pb2, h2, Nctx, H_DIM);
        gemm_wmma<0, 1><<<g, blk, 0, stream>>>(h2, H_DIM, nullptr, H_DIM, pw3, pb3, pe, Nctx, LD_DIM);
    }

    // 2) gather ctx = pair_embed[idx]
    {
        const size_t total = (size_t)BL * LD_DIM;
        gather_ctx<<<dim3((unsigned)((total + 255) / 256)), blk, 0, stream>>>(pe, sse, ctx, Nctx);
    }

    // 3) q/k/v projections
    {
        const dim3 g(LD_DIM / 64, BL / 128);
        gemm_wmma<0, 1><<<g, blk, 0, stream>>>(ctx, LD_DIM, nullptr, LD_DIM, wq, bq, qb, BL, LD_DIM);
        gemm_wmma<0, 1><<<g, blk, 0, stream>>>(ctx, LD_DIM, nullptr, LD_DIM, wk, bk, kb, BL, LD_DIM);
        gemm_wmma<0, 1><<<g, blk, 0, stream>>>(ctx, LD_DIM, nullptr, LD_DIM, wv, bv, vb, BL, LD_DIM);
    }

    // 4) attention + masked mean pooling -> seg_out
    attn_seg<<<dim3(B_DIM), blk, 0, stream>>>(qb, kb, vb, sse, seg);

    // 5) mean / log_var (clipped), written straight into d_out
    {
        const dim3 g(LD_DIM / 64, B_DIM / 128);
        gemm_wmma<2, 1><<<g, blk, 0, stream>>>(seg, LD_DIM, nullptr, LD_DIM, wm, bm, omean, B_DIM, LD_DIM);
        gemm_wmma<2, 1><<<g, blk, 0, stream>>>(seg, LD_DIM, nullptr, LD_DIM, wlv, blv, olv, B_DIM, LD_DIM);
    }

    // 6) reparameterize + normalize -> z
    z_kernel<<<dim3(B_DIM), blk, 0, stream>>>(omean, olv, eps, oz);

    // 7) decoder, chosen then rejected
    {
        const dim3 g(H_DIM / 64, B_DIM / 128);
        gemm_wmma<1, 0><<<g, blk, 0, stream>>>(tc, E_DIM, oz, E_DIM + LD_DIM, dw1, db1, u1, B_DIM, H_DIM);
        gemm_wmma<1, 1><<<g, blk, 0, stream>>>(u1, H_DIM, nullptr, H_DIM, dw2, db2, u2, B_DIM, H_DIM);
        rowdot<<<dim3(B_DIM / 8), blk, 0, stream>>>(u2, dw3, db3, rc, B_DIM);

        gemm_wmma<1, 0><<<g, blk, 0, stream>>>(tr, E_DIM, oz, E_DIM + LD_DIM, dw1, db1, u1, B_DIM, H_DIM);
        gemm_wmma<1, 1><<<g, blk, 0, stream>>>(u1, H_DIM, nullptr, H_DIM, dw2, db2, u2, B_DIM, H_DIM);
        rowdot<<<dim3(B_DIM / 8), blk, 0, stream>>>(u2, dw3, db3, rr, B_DIM);
    }
}
